// Model_20547123544401
// MI455X (gfx1250) — compile-verified, matrix operand order checked by reading, not run on hardware
//
#include <hip/hip_runtime.h>
#include <hip/hip_bf16.h>
#include <cstdint>

typedef __attribute__((ext_vector_type(16))) _Float16 v16h;
typedef __attribute__((ext_vector_type(8)))  _Float16 v8h;
typedef __attribute__((ext_vector_type(8)))  float    v8f;
typedef __attribute__((ext_vector_type(4)))  unsigned int v4u;
typedef __attribute__((ext_vector_type(8)))  int      v8i;
typedef __attribute__((ext_vector_type(4)))  int      v4i;

#define B_  32
#define T_  512
#define C_  768
#define H_  8
#define HS_ 96
#define M_  (B_ * T_)          // 16384 tokens
#define N3_ (3 * C_)           // 2304
#define LDS_PITCH 40           // halves per staged row (32 data + 8 pad)
#define BUF_HALVES (32 * LDS_PITCH)
static __device__ __constant__ float kQScale = 0.10206207261596575f; // 1/sqrt(96)

// ---------------------------------------------------------------------------
// Fragment loader from global (contiguous K): lane l (m=l&15, half=l>>4) holds
// row row0+m, kk = {half*8..+7} and {16+half*8..+7} -> two b128 loads.
// ---------------------------------------------------------------------------
__device__ inline v16h load_frag(const _Float16* base, int row0, int stride, int col0) {
    const int l = threadIdx.x & 31, m = l & 15, half = l >> 4;
    const _Float16* p = base + (size_t)(row0 + m) * stride + col0 + half * 8;
    v8h lo = *(const v8h*)(p);
    v8h hi = *(const v8h*)(p + 16);
    v16h a;
#pragma unroll
    for (int i = 0; i < 8; ++i) { a[i] = lo[i]; a[i + 8] = hi[i]; }
    return a;
}

// Same fragment pattern out of a TDM-padded LDS tile (pitch 40 halves).
__device__ inline v16h lds_frag(const _Float16* base, int row0) {
    const int l = threadIdx.x & 31, m = l & 15, half = l >> 4;
    const _Float16* p = base + (row0 + m) * LDS_PITCH + half * 8;
    v8h lo = *(const v8h*)(p);
    v8h hi = *(const v8h*)(p + 16);
    v16h a;
#pragma unroll
    for (int i = 0; i < 8; ++i) { a[i] = lo[i]; a[i + 8] = hi[i]; }
    return a;
}

__device__ inline v8f wmma16(v16h a, v16h b, v8f c) {
    return __builtin_amdgcn_wmma_f32_16x16x32_f16(false, a, false, b, (short)0, c, false, false);
}

__device__ inline float rmax16(float v) {
#pragma unroll
    for (int o = 1; o < 16; o <<= 1) v = fmaxf(v, __shfl_xor(v, o, 32));
    return v;
}
__device__ inline float rsum16(float v) {
#pragma unroll
    for (int o = 1; o < 16; o <<= 1) v += __shfl_xor(v, o, 32);
    return v;
}

// ---------------------------------------------------------------------------
// TDM: stage a 32x32 f16 tile (row stride src_stride elements) into LDS with
// 16B row padding (pitch 40 halves). D# per CDNA5 ISA ch.8; uniform -> SGPRs.
// ---------------------------------------------------------------------------
__device__ inline void tdm_load_tile_32x32(const _Float16* src, int src_stride,
                                           _Float16* ldsdst) {
    uint64_t ga = (uint64_t)(uintptr_t)src;
    unsigned int lds_off = (unsigned int)(uintptr_t)ldsdst; // low 32b = LDS byte offset
    v4u g0;
    g0[0] = (unsigned)__builtin_amdgcn_readfirstlane(1);                 // count=1
    g0[1] = (unsigned)__builtin_amdgcn_readfirstlane((int)lds_off);      // lds_addr
    g0[2] = (unsigned)__builtin_amdgcn_readfirstlane((int)(unsigned)ga); // gaddr[31:0]
    g0[3] = (unsigned)__builtin_amdgcn_readfirstlane(
        (int)(((unsigned)(ga >> 32) & 0x1FFFFFFu) | (2u << 30)));        // gaddr[56:32] | type=2
    v8i g1;
    // data_size=1(2B) | pad_enable | pad_interval=3(16 DW) | pad_amount=3(4 DW)
    g1[0] = __builtin_amdgcn_readfirstlane(
        (int)(0x10000u | (1u << 20) | (3u << 22) | (3u << 25)));
    g1[1] = __builtin_amdgcn_readfirstlane((int)(32u << 16)); // tensor_dim0 lo16 = 32
    g1[2] = __builtin_amdgcn_readfirstlane((int)(32u << 16)); // dim0 hi=0 | tensor_dim1 lo16 = 32
    g1[3] = __builtin_amdgcn_readfirstlane((int)(32u << 16)); // dim1 hi=0 | tile_dim0 = 32
    g1[4] = __builtin_amdgcn_readfirstlane(32);               // tile_dim1 = 32, tile_dim2 = 0
    g1[5] = __builtin_amdgcn_readfirstlane(src_stride);       // tensor_dim0_stride lo32
    g1[6] = 0;
    g1[7] = 0;
    v4i z4 = {0, 0, 0, 0};
    v8i z8 = {0, 0, 0, 0, 0, 0, 0, 0};
    __builtin_amdgcn_tensor_load_to_lds(g0, g1, z4, z4, z8, 0);
}

// ------------------------------ f32 -> f16 convert -------------------------
__global__ __launch_bounds__(256) void k_conv(const float* __restrict__ src,
                                              _Float16* __restrict__ dst, int n) {
    int id = blockIdx.x * 256 + threadIdx.x;
    if (id < n) dst[id] = (_Float16)src[id];
}

// transpose-convert: src is R x Cc (f32, row-major) -> dst is Cc x R (f16)
__global__ __launch_bounds__(256) void k_tconv(const float* __restrict__ src,
                                               _Float16* __restrict__ dst, int R, int Cc) {
    int id = blockIdx.x * 256 + threadIdx.x;
    if (id < R * Cc) {
        int r = id % R, c = id / R;
        dst[id] = (_Float16)src[(size_t)r * Cc + c];
    }
}

// ---------------------------------------------------------------------------
// 32x32-per-wave GEMM mainloop: A frags direct from global, B tile staged in
// LDS by the Tensor Data Mover (double-buffered on TENSORcnt).
// ---------------------------------------------------------------------------
__device__ inline void gemm32_core(const _Float16* __restrict__ A, int lda, int row_a0,
                                   const _Float16* __restrict__ Brows, int ldb,
                                   _Float16* lds, v8f acc[4]) {
    const int STEPS = lda / 32; // K assumed == lda here (full-row reduction)
    tdm_load_tile_32x32(Brows, ldb, lds);
    for (int s = 0; s < STEPS; ++s) {
        const int k0 = s * 32;
        if (s + 1 < STEPS) {
            tdm_load_tile_32x32(Brows + k0 + 32, ldb, lds + ((s + 1) & 1) * BUF_HALVES);
            __builtin_amdgcn_s_wait_tensorcnt((short)1);
        } else {
            __builtin_amdgcn_s_wait_tensorcnt((short)0);
        }
        asm volatile("" ::: "memory");   // keep LDS reads below the tensorcnt wait
        v16h a0 = load_frag(A, row_a0,      lda, k0);
        v16h a1 = load_frag(A, row_a0 + 16, lda, k0);
        const _Float16* bb = lds + (s & 1) * BUF_HALVES;
        v16h b0 = lds_frag(bb, 0);
        v16h b1 = lds_frag(bb, 16);
        acc[0] = wmma16(a0, b0, acc[0]);
        acc[1] = wmma16(a0, b1, acc[1]);
        acc[2] = wmma16(a1, b0, acc[2]);
        acc[3] = wmma16(a1, b1, acc[3]);
    }
}

// ------------------------------ QKV GEMM -----------------------------------
// qkv[m][n] = sum_c xh[m][c] * WqkvT[n][c] + bqkv[n]
// Q (scaled) -> qh[B,H,T,hs], K -> kh[B,H,T,hs], V -> vth[B,H,hs,T]
__global__ __launch_bounds__(256) void k_qkv_gemm(const _Float16* __restrict__ xh,
                                                  const _Float16* __restrict__ wT,
                                                  const float* __restrict__ bias,
                                                  _Float16* __restrict__ qh,
                                                  _Float16* __restrict__ kh,
                                                  _Float16* __restrict__ vth) {
    __shared__ _Float16 Bstage[8][2 * BUF_HALVES];
    const int wid  = threadIdx.x >> 5;
    const int wave = blockIdx.x * 8 + wid;
    const int TN = N3_ / 32;                 // 72 column tiles
    const int tm = wave / TN, tn = wave % TN;
    v8f acc[4];
#pragma unroll
    for (int j = 0; j < 4; ++j) acc[j] = (v8f){};
    gemm32_core(xh, C_, tm * 32, wT + (size_t)tn * 32 * C_, C_, &Bstage[wid][0], acc);

    const int l = threadIdx.x & 31, m = l & 15, half = l >> 4;
    const int region = tn / 24;              // 0:Q 1:K 2:V (32-col tile never crosses)
#pragma unroll
    for (int in = 0; in < 2; ++in) {
        const int n = tn * 32 + in * 16 + m;
        const float bv = bias[n];
        const int n2 = n - region * C_;
        const int h = n2 / HS_, d = n2 % HS_;
#pragma unroll
        for (int im = 0; im < 2; ++im) {
            v8f cc = acc[im * 2 + in];
#pragma unroll
            for (int i = 0; i < 8; ++i) {
                int row = tm * 32 + im * 16 + i + 8 * half;
                int b_ = row >> 9, t = row & (T_ - 1);
                float v = cc[i] + bv;
                if (region == 0) {
                    qh[(((size_t)(b_ * H_ + h) * T_) + t) * HS_ + d] = (_Float16)(v * kQScale);
                } else if (region == 1) {
                    kh[(((size_t)(b_ * H_ + h) * T_) + t) * HS_ + d] = (_Float16)v;
                } else {
                    vth[(((size_t)(b_ * H_ + h) * HS_) + d) * T_ + t] = (_Float16)v;
                }
            }
        }
    }
}

// ------------------------------ flash attention ----------------------------
__global__ __launch_bounds__(256) void k_attn(const _Float16* __restrict__ qh,
                                              const _Float16* __restrict__ kh,
                                              const _Float16* __restrict__ vth,
                                              _Float16* __restrict__ yh) {
    __shared__ _Float16 Plds[8][16 * 40];
    const int wid  = threadIdx.x >> 5;
    const int wave = blockIdx.x * 8 + wid;   // 0 .. B*H*(T/16)-1
    const int qt = wave & 31;
    const int bh = wave >> 5;
    const int h = bh & (H_ - 1), b_ = bh >> 3;
    const _Float16* qb = qh  + (size_t)bh * T_ * HS_;
    const _Float16* kb = kh  + (size_t)bh * T_ * HS_;
    const _Float16* vb = vth + (size_t)bh * HS_ * T_;
    _Float16* P = &Plds[wid][0];

    const int q0 = qt * 16;
    const int l = threadIdx.x & 31, m = l & 15, half = l >> 4;

    v16h qa[3];
#pragma unroll
    for (int kd = 0; kd < 3; ++kd) qa[kd] = load_frag(qb, q0, HS_, kd * 32);

    v8f acc[6];
    float ms[8], ls[8];
#pragma unroll
    for (int t = 0; t < 6; ++t) acc[t] = (v8f){};
#pragma unroll
    for (int i = 0; i < 8; ++i) { ms[i] = -__builtin_inff(); ls[i] = 0.0f; }

    for (int j0 = 0; j0 <= qt; j0 += 2) {    // key-tile pair (2nd may be fully masked)
        const int key0 = j0 * 16;
        v8f s0 = {}, s1 = {};
#pragma unroll
        for (int kd = 0; kd < 3; ++kd) {
            v16h bk0 = load_frag(kb, key0,      HS_, kd * 32);
            v16h bk1 = load_frag(kb, key0 + 16, HS_, kd * 32);
            s0 = wmma16(qa[kd], bk0, s0);
            s1 = wmma16(qa[kd], bk1, s1);
        }
#pragma unroll
        for (int i = 0; i < 8; ++i) {
            int row = q0 + i + 8 * half;
            float v0 = (key0 + m      <= row) ? s0[i] : -__builtin_inff();
            float v1 = (key0 + 16 + m <= row) ? s1[i] : -__builtin_inff();
            float tmax  = rmax16(fmaxf(v0, v1));
            float mnew  = fmaxf(ms[i], tmax);
            float alpha = __expf(ms[i] - mnew);
            float p0 = __expf(v0 - mnew);
            float p1 = __expf(v1 - mnew);
            float rs = rsum16(p0 + p1);
            ls[i] = ls[i] * alpha + rs;
            ms[i] = mnew;
#pragma unroll
            for (int t = 0; t < 6; ++t) acc[t][i] *= alpha;
            P[(i + 8 * half) * 40 + m]      = (_Float16)p0;
            P[(i + 8 * half) * 40 + m + 16] = (_Float16)p1;
        }
        asm volatile("s_wait_dscnt 0x0" ::: "memory");   // in-wave LDS transpose handoff
        v16h pa;
        {
            const _Float16* pp = P + m * 40 + half * 8;
            v8h lo = *(const v8h*)(pp);
            v8h hi = *(const v8h*)(pp + 16);
#pragma unroll
            for (int i = 0; i < 8; ++i) { pa[i] = lo[i]; pa[i + 8] = hi[i]; }
        }
#pragma unroll
        for (int t = 0; t < 6; ++t) {
            v16h vf = load_frag(vb, t * 16, T_, key0);
            acc[t] = wmma16(pa, vf, acc[t]);
        }
    }
#pragma unroll
    for (int i = 0; i < 8; ++i) {
        float inv = 1.0f / ls[i];
        int tok = q0 + i + 8 * half;
        size_t rowbase = ((size_t)b_ * T_ + tok) * C_ + h * HS_;
#pragma unroll
        for (int t = 0; t < 6; ++t)
            yh[rowbase + t * 16 + m] = (_Float16)(acc[t][i] * inv);
    }
}

// ------------------------------ output projection --------------------------
__global__ __launch_bounds__(256) void k_proj_gemm(const _Float16* __restrict__ yh,
                                                   const _Float16* __restrict__ wT,
                                                   const float* __restrict__ bias,
                                                   float* __restrict__ out) {
    __shared__ _Float16 Bstage[8][2 * BUF_HALVES];
    const int wid  = threadIdx.x >> 5;
    const int wave = blockIdx.x * 8 + wid;
    const int TN = C_ / 32;                  // 24
    const int tm = wave / TN, tn = wave % TN;
    v8f acc[4];
#pragma unroll
    for (int j = 0; j < 4; ++j) acc[j] = (v8f){};
    gemm32_core(yh, C_, tm * 32, wT + (size_t)tn * 32 * C_, C_, &Bstage[wid][0], acc);

    const int l = threadIdx.x & 31, m = l & 15, half = l >> 4;
#pragma unroll
    for (int in = 0; in < 2; ++in) {
        const int n = tn * 32 + in * 16 + m;
        const float bv = bias[n];
#pragma unroll
        for (int im = 0; im < 2; ++im) {
            v8f cc = acc[im * 2 + in];
#pragma unroll
            for (int i = 0; i < 8; ++i) {
                int row = tm * 32 + im * 16 + i + 8 * half;
                out[(size_t)row * C_ + n] = cc[i] + bv;
            }
        }
    }
}

// ---------------------------------------------------------------------------
extern "C" void kernel_launch(void* const* d_in, const int* in_sizes, int n_in,
                              void* d_out, int out_size, void* d_ws, size_t ws_size,
                              hipStream_t stream) {
    const float* x     = (const float*)d_in[0];   // [B,T,C]
    const float* Wqkv  = (const float*)d_in[1];   // [C,3C]
    const float* bqkv  = (const float*)d_in[2];   // [3C]
    const float* Wproj = (const float*)d_in[3];   // [C,C]
    const float* bproj = (const float*)d_in[4];   // [C]
    float* out = (float*)d_out;

    char* ws = (char*)d_ws;
    const size_t SZ_MC = (size_t)M_ * C_ * sizeof(_Float16);       // 25,165,824 B
    _Float16* xh  = (_Float16*)(ws);
    _Float16* qh  = (_Float16*)(ws + SZ_MC);
    _Float16* kh  = (_Float16*)(ws + 2 * SZ_MC);
    _Float16* vth = (_Float16*)(ws + 3 * SZ_MC);
    _Float16* yh  = (_Float16*)(ws + 4 * SZ_MC);
    _Float16* wqT = (_Float16*)(ws + 5 * SZ_MC);                   // [3C][C]
    _Float16* wpT = (_Float16*)(ws + 5 * SZ_MC + (size_t)N3_ * C_ * sizeof(_Float16)); // [C][C]

    k_conv <<<(M_ * C_) / 256, 256, 0, stream>>>(x, xh, M_ * C_);
    k_tconv<<<(C_ * N3_) / 256, 256, 0, stream>>>(Wqkv, wqT, C_, N3_);
    k_tconv<<<(C_ * C_)  / 256, 256, 0, stream>>>(Wproj, wpT, C_, C_);

    {
        int waves = (M_ / 32) * (N3_ / 32);      // 36864
        k_qkv_gemm<<<waves / 8, 256, 0, stream>>>(xh, wqT, bqkv, qh, kh, vth);
    }
    {
        int waves = B_ * H_ * (T_ / 16);         // 8192
        k_attn<<<waves / 8, 256, 0, stream>>>(qh, kh, vth, yh);
    }
    {
        int waves = (M_ / 32) * (C_ / 32);       // 12288
        k_proj_gemm<<<waves / 8, 256, 0, stream>>>(yh, wpT, bproj, out);
    }
    (void)in_sizes; (void)n_in; (void)out_size; (void)ws_size;
}
